// TeacherForcingDecoder_72421738545905
// MI455X (gfx1250) — matrix operand order repaired and checked by reading, not compiled, not run on hardware
//
#include <hip/hip_runtime.h>
#include <hip/hip_bf16.h>

// ---------------------------------------------------------------------------
// TeacherForcingDecoder fused GRU-step + LayerNorm for MI455X (gfx1250).
// Dominant GEMM reduced 4x by pre-summing w_hh's tiled columns; all GEMMs use
// v_wmma_f32_16x16x32_bf16 with bf16x3 split (hi*hi + hi*lo + lo*hi) for
// ~fp32 accuracy. Kernel is HBM-bound (~300MB traffic -> ~13us floor).
// ---------------------------------------------------------------------------

typedef __attribute__((ext_vector_type(16))) __bf16 v16bf;
typedef __attribute__((ext_vector_type(8)))  float  v8f;

#define T_ 28
#define B_ 2048
#define L_ 512
#define E_ 256
#define D_ 1024
#define G_ 3072   // 3*D

__device__ __forceinline__ v8f wmma_bf16(v16bf a, v16bf b, v8f c) {
  return __builtin_amdgcn_wmma_f32_16x16x32_bf16(
      /*neg_a=*/false, a, /*neg_b=*/false, b,
      /*c_mod=*/(short)0, c, /*reuse_a=*/false, /*reuse_b=*/false);
}

// Build A fragment (16x32 bf16, hi+lo split) for this lane from an f32 row.
// A layout: lane m (both halves) holds row m; lanes 0-15 K{0-15},
// lanes 16-31 K{16-31}.
__device__ __forceinline__ void load_a_frag(const float* __restrict__ rowptr,
                                            int k0a, int k0b,
                                            v16bf& ahi, v16bf& alo) {
  float f[16];
  if (rowptr) {
#pragma unroll
    for (int j = 0; j < 8; ++j) f[j] = rowptr[k0a + j];
#pragma unroll
    for (int j = 0; j < 8; ++j) f[8 + j] = rowptr[k0b + j];
  } else {
#pragma unroll
    for (int j = 0; j < 16; ++j) f[j] = 0.0f;
  }
#pragma unroll
  for (int j = 0; j < 16; ++j) {
    unsigned u = __builtin_bit_cast(unsigned, f[j]) & 0xFFFF0000u;
    float hi = __builtin_bit_cast(float, u);
    ahi[j] = (__bf16)hi;
    alo[j] = (__bf16)(f[j] - hi);
  }
}

// B fragment (32x16): lane holds 16 consecutive K of column n = lane&15.
// W is row-major [N x K] bf16, so this is a straight 32B contiguous load.
__device__ __forceinline__ v16bf load_b_frag(const __bf16* __restrict__ W,
                                             int K, int n, int kbase) {
  return *reinterpret_cast<const v16bf*>(W + (size_t)n * K + kbase);
}

__device__ __forceinline__ float sigmoid_f(float x) {
  return 1.0f / (1.0f + __expf(-x));
}
__device__ __forceinline__ float tanh_f(float x) {
  return 1.0f - 2.0f / (__expf(2.0f * x) + 1.0f);
}

// ---------------------------------------------------------------------------
// Weight pack: optional 4-way column-block reduce, then f32 -> bf16 hi/lo.
// ---------------------------------------------------------------------------
__global__ void pack_w_kernel(const float* __restrict__ src,
                              __bf16* __restrict__ hi, __bf16* __restrict__ lo,
                              int N, int Kd, int reduce4) {
  size_t i = (size_t)blockIdx.x * blockDim.x + threadIdx.x;
  if (i >= (size_t)N * Kd) return;
  int n = (int)(i / Kd), k = (int)(i % Kd);
  float v;
  if (reduce4) {
    const float* r = src + (size_t)n * (4 * Kd);
    v = r[k] + r[k + Kd] + r[k + 2 * Kd] + r[k + 3 * Kd];
  } else {
    v = src[(size_t)n * Kd + k];
  }
  unsigned u = __builtin_bit_cast(unsigned, v) & 0xFFFF0000u;
  float h = __builtin_bit_cast(float, u);
  hi[i] = (__bf16)h;
  lo[i] = (__bf16)(v - h);
}

// ---------------------------------------------------------------------------
// Generic C = A @ B^T + bias, bf16x3 WMMA. A: f32 [M x K]; Bhi/Blo: bf16
// [N x K]; block = 16 rows x (waves * NT * 16) cols.
// ---------------------------------------------------------------------------
template <int NT>
__global__ __launch_bounds__(256) void gemm_bf16x3_kernel(
    const float* __restrict__ A, const __bf16* __restrict__ Bhi,
    const __bf16* __restrict__ Blo, const float* __restrict__ bias,
    float* __restrict__ C, int M, int N, int K) {
  const int lane = threadIdx.x & 31;
  const int wave = threadIdx.x >> 5;
  const int waves = blockDim.x >> 5;
  const int l15 = lane & 15;
  const int lhalf = lane >> 4;
  const int row0 = blockIdx.x * 16;
  const int ntile0 = (blockIdx.y * waves + wave) * NT;

  v8f acc[NT];
#pragma unroll
  for (int t = 0; t < NT; ++t) {
    float b = bias[(ntile0 + t) * 16 + l15];
#pragma unroll
    for (int v = 0; v < 8; ++v) acc[t][v] = b;
  }

  const float* arow = A + (size_t)(row0 + l15) * K;
  const int nks = K >> 5;
  for (int ks = 0; ks < nks; ++ks) {
    const int k0a = ks * 32 + lhalf * 16;
    v16bf ahi, alo;
    load_a_frag(arow, k0a, k0a + 8, ahi, alo);
    const int kb = ks * 32 + lhalf * 16;
#pragma unroll
    for (int t = 0; t < NT; ++t) {
      const int n = (ntile0 + t) * 16 + l15;
      v16bf bh = load_b_frag(Bhi, K, n, kb);
      v16bf bl = load_b_frag(Blo, K, n, kb);
      acc[t] = wmma_bf16(ahi, bh, acc[t]);
      acc[t] = wmma_bf16(ahi, bl, acc[t]);
      acc[t] = wmma_bf16(alo, bh, acc[t]);
    }
  }

#pragma unroll
  for (int t = 0; t < NT; ++t) {
    const int col = (ntile0 + t) * 16 + l15;
#pragma unroll
    for (int v = 0; v < 8; ++v) {
      const int row = row0 + v + lhalf * 8;
      C[(size_t)row * N + col] = acc[t][v];
    }
  }
}

// ---------------------------------------------------------------------------
// Fused: gh GEMM (K=256, reduced weights) + GRU gates + LayerNorm + store.
// Block: 16 flattened (t,b) rows x 1024 cols; 8 waves x 8 d-tiles each.
// ---------------------------------------------------------------------------
__global__ __launch_bounds__(256) void decoder_main_kernel(
    const float* __restrict__ true_inp,   // [B, T, E]
    const float* __restrict__ gi,         // [T? no: B rows replicated via t] -> [TB? ] see below
    const __bf16* __restrict__ Whi,       // [3072 x 256] reduced w_hh hi
    const __bf16* __restrict__ Wlo,       // lo
    const float* __restrict__ b_hh,       // [3072]
    const float* __restrict__ gamma, const float* __restrict__ beta,
    float* __restrict__ out) {            // [T*B, D]
  __shared__ float s_sum[16];
  __shared__ float s_ssq[16];

  const int lane = threadIdx.x & 31;
  const int wave = threadIdx.x >> 5;  // 0..7
  const int l15 = lane & 15;
  const int lhalf = lane >> 4;
  const int row0 = blockIdx.x * 16;   // flattened r = t*B + b (t constant per tile)
  const int t = row0 / B_;
  const int b0 = row0 % B_;

  if (threadIdx.x < 16) { s_sum[threadIdx.x] = 0.0f; s_ssq[threadIdx.x] = 0.0f; }
  __syncthreads();

  // A = shifted teacher row (zeros for t==0), kept fully in registers (K=256).
  const float* arow =
      (t == 0) ? nullptr
               : (true_inp + ((size_t)(b0 + l15) * T_ + (t - 1)) * E_);
  v16bf ahi[8], alo[8];
#pragma unroll
  for (int ks = 0; ks < 8; ++ks) {
    const int k0a = ks * 32 + lhalf * 16;
    load_a_frag(arow, k0a, k0a + 8, ahi[ks], alo[ks]);
  }

  float outv[8][8];
  float rsum[8], rssq[8];
#pragma unroll
  for (int v = 0; v < 8; ++v) { rsum[v] = 0.0f; rssq[v] = 0.0f; }

#pragma unroll 1
  for (int dt = 0; dt < 8; ++dt) {
    const int dtile = wave * 8 + dt;         // 0..63
    const int dcol = dtile * 16 + l15;       // 0..1023 (this lane's column)

    v8f accR, accZ, accN;
    {
      const float bR = b_hh[dcol];
      const float bZ = b_hh[D_ + dcol];
      const float bN = b_hh[2 * D_ + dcol];
#pragma unroll
      for (int v = 0; v < 8; ++v) { accR[v] = bR; accZ[v] = bZ; accN[v] = bN; }
    }
    // r,z gates get gi added up-front; n gate keeps hidden part separate
    // because reference computes tanh(i_n + r * h_n).
#pragma unroll
    for (int v = 0; v < 8; ++v) {
      const size_t row = (size_t)(b0 + v + lhalf * 8);
      accR[v] += gi[row * G_ + dcol];
      accZ[v] += gi[row * G_ + D_ + dcol];
    }

    if (t != 0) {
#pragma unroll
      for (int ks = 0; ks < 8; ++ks) {
        const int kb = ks * 32 + lhalf * 16;
        v16bf bh, bl;
        bh = load_b_frag(Whi, E_, 0 * D_ + dcol, kb);
        bl = load_b_frag(Wlo, E_, 0 * D_ + dcol, kb);
        accR = wmma_bf16(ahi[ks], bh, accR);
        accR = wmma_bf16(ahi[ks], bl, accR);
        accR = wmma_bf16(alo[ks], bh, accR);
        bh = load_b_frag(Whi, E_, 1 * D_ + dcol, kb);
        bl = load_b_frag(Wlo, E_, 1 * D_ + dcol, kb);
        accZ = wmma_bf16(ahi[ks], bh, accZ);
        accZ = wmma_bf16(ahi[ks], bl, accZ);
        accZ = wmma_bf16(alo[ks], bh, accZ);
        bh = load_b_frag(Whi, E_, 2 * D_ + dcol, kb);
        bl = load_b_frag(Wlo, E_, 2 * D_ + dcol, kb);
        accN = wmma_bf16(ahi[ks], bh, accN);
        accN = wmma_bf16(ahi[ks], bl, accN);
        accN = wmma_bf16(alo[ks], bh, accN);
      }
    }

    // gates + blend, keep pre-LN values in registers
#pragma unroll
    for (int v = 0; v < 8; ++v) {
      const int b = b0 + v + lhalf * 8;
      const float gin = gi[(size_t)b * G_ + 2 * D_ + dcol];
      const float hv =
          (t == 0) ? 0.0f
                   : true_inp[((size_t)b * T_ + (t - 1)) * E_ + (dcol & (E_ - 1))];
      const float r = sigmoid_f(accR[v]);
      const float u = sigmoid_f(accZ[v]);
      const float n = tanh_f(gin + r * accN[v]);
      const float o = (1.0f - u) * n + u * hv;
      outv[dt][v] = o;
      rsum[v] += o;
      rssq[v] += o * o;
    }
  }

  // LayerNorm row statistics across the whole workgroup (row = v + lhalf*8)
#pragma unroll
  for (int v = 0; v < 8; ++v) {
    atomicAdd(&s_sum[v + lhalf * 8], rsum[v]);
    atomicAdd(&s_ssq[v + lhalf * 8], rssq[v]);
  }
  __syncthreads();

#pragma unroll
  for (int v = 0; v < 8; ++v) {
    const int m = v + lhalf * 8;
    const float mu = s_sum[m] * (1.0f / (float)D_);
    const float var = s_ssq[m] * (1.0f / (float)D_) - mu * mu;
    const float rs = rsqrtf(var + 1e-5f);
    const size_t row = (size_t)(row0 + m);
#pragma unroll
    for (int dt = 0; dt < 8; ++dt) {
      const int dcol = (wave * 8 + dt) * 16 + l15;
      out[row * D_ + dcol] = (outv[dt][v] - mu) * rs * gamma[dcol] + beta[dcol];
    }
  }
}

// ---------------------------------------------------------------------------
extern "C" void kernel_launch(void* const* d_in, const int* in_sizes, int n_in,
                              void* d_out, int out_size, void* d_ws,
                              size_t ws_size, hipStream_t stream) {
  const float* z        = (const float*)d_in[0];   // [B, L]
  const float* true_inp = (const float*)d_in[1];   // [B, T, E]
  // d_in[2] = tf_prob (unused: reference ignores it, states are pure teacher)
  const float* fc_w     = (const float*)d_in[3];   // [D, L]
  const float* fc_b     = (const float*)d_in[4];   // [D]
  const float* w_ih     = (const float*)d_in[5];   // [3D, D]
  const float* w_hh     = (const float*)d_in[6];   // [3D, D]
  const float* b_ih     = (const float*)d_in[7];   // [3D]
  const float* b_hh     = (const float*)d_in[8];   // [3D]
  const float* ln_gamma = (const float*)d_in[9];   // [D]
  const float* ln_beta  = (const float*)d_in[10];  // [D]
  float* out = (float*)d_out;                      // [T, B, D]

  // Workspace layout (bytes, 256-aligned)
  char* ws = (char*)d_ws;
  size_t off = 0;
  auto alloc = [&](size_t bytes) {
    char* p = ws + off;
    off = (off + bytes + 255) & ~(size_t)255;
    return p;
  };
  __bf16* whh_hi = (__bf16*)alloc((size_t)G_ * E_ * 2);   // 1.5 MB
  __bf16* whh_lo = (__bf16*)alloc((size_t)G_ * E_ * 2);
  __bf16* fcw_hi = (__bf16*)alloc((size_t)D_ * L_ * 2);   // 1 MB
  __bf16* fcw_lo = (__bf16*)alloc((size_t)D_ * L_ * 2);
  __bf16* wih_hi = (__bf16*)alloc((size_t)G_ * D_ * 2);   // 6 MB
  __bf16* wih_lo = (__bf16*)alloc((size_t)G_ * D_ * 2);
  float*  x_buf  = (float*)alloc((size_t)B_ * D_ * 4);    // 8 MB
  float*  gi_buf = (float*)alloc((size_t)B_ * G_ * 4);    // 24 MB
  (void)ws_size; (void)in_sizes; (void)n_in; (void)out_size;

  // 1) Pack weights to bf16 hi/lo (w_hh gets the 4-way column-block reduce).
  {
    int n = D_ * L_;
    pack_w_kernel<<<(n + 255) / 256, 256, 0, stream>>>(fc_w, fcw_hi, fcw_lo,
                                                       D_, L_, 0);
  }
  {
    int n = G_ * D_;
    pack_w_kernel<<<(n + 255) / 256, 256, 0, stream>>>(w_ih, wih_hi, wih_lo,
                                                       G_, D_, 0);
  }
  {
    int n = G_ * E_;
    pack_w_kernel<<<(n + 255) / 256, 256, 0, stream>>>(w_hh, whh_hi, whh_lo,
                                                       G_, E_, 1);
  }

  // 2) x = z @ fc_w^T + fc_b   [2048 x 1024], K=512
  gemm_bf16x3_kernel<2><<<dim3(B_ / 16, D_ / 256), 256, 0, stream>>>(
      z, fcw_hi, fcw_lo, fc_b, x_buf, B_, D_, L_);

  // 3) gi = x @ w_ih^T + b_ih  [2048 x 3072], K=1024
  gemm_bf16x3_kernel<2><<<dim3(B_ / 16, G_ / 256), 256, 0, stream>>>(
      x_buf, wih_hi, wih_lo, b_ih, gi_buf, B_, G_, D_);

  // 4) Fused gh GEMM + gates + LayerNorm over all T*B rows.
  decoder_main_kernel<<<(T_ * B_) / 16, 256, 0, stream>>>(
      true_inp, gi_buf, whh_hi, whh_lo, b_hh, ln_gamma, ln_beta, out);
}
